// GraphAttentionLayer_30150670418133
// MI455X (gfx1250) — matrix operand order repaired and checked by reading, not compiled, not run on hardware
//
#include <hip/hip_runtime.h>
#include <math.h>

// CDNA5 (gfx1250) wave32 WMMA types
typedef __attribute__((ext_vector_type(16))) _Float16 v16h;
typedef __attribute__((ext_vector_type(8)))  _Float16 v8h;
typedef __attribute__((ext_vector_type(8)))  float    v8f;
typedef __attribute__((ext_vector_type(4)))  int      v4i;

#define N_ROWS 8192
#define DIN    256
#define DOUT   128
#define KSPLIT 4            // key-dimension split per query tile
#define KSPAN  (N_ROWS / KSPLIT)

__device__ __forceinline__ v8f wmma_f16(v16h a, v16h b, v8f c) {
  // D = A(16x32 f16) * B(32x16 f16) + C(16x16 f32)
  return __builtin_amdgcn_wmma_f32_16x16x32_f16(
      /*neg_a=*/false, a, /*neg_b=*/false, b,
      /*c_mod=*/(short)0, c, /*reuse_a=*/false, /*reuse_b=*/false);
}

// ---------------------------------------------------------------------------
// Kernel 1: QKV projection (x[8192,256] f32 @ W[256,128] f32 + b) -> f16.
//   mat 0: Qh[row][d]   pre-scaled by log2(e)/sqrt(128) (bias included) so the
//                       attention loop can use raw v_exp_f32 (base-2).
//   mat 1: Kh[row][d]
//   mat 2: Vt[d][row]   transposed so P@V B-fragments are contiguous loads
// One wave = one 16-row strip of one matrix. 1536 wave-jobs.
// ---------------------------------------------------------------------------
__global__ void __launch_bounds__(256) qkv_proj_kernel(
    const float* __restrict__ x,
    const float* __restrict__ Wq, const float* __restrict__ bq,
    const float* __restrict__ Wk, const float* __restrict__ bk,
    const float* __restrict__ Wv, const float* __restrict__ bv,
    _Float16* __restrict__ Qh, _Float16* __restrict__ Kh,
    _Float16* __restrict__ Vt)
{
  const int lane = threadIdx.x & 31;
  const int wid  = threadIdx.x >> 5;
  const int job  = blockIdx.x * 8 + wid;   // 0..1535
  const int mat  = job >> 9;               // 0,1,2
  const int tile = job & 511;              // 16-row tile
  const int hi   = lane >> 4;
  const int ln   = lane & 15;

  const float* W;
  const float* bias;
  if (mat == 0)      { W = Wq; bias = bq; }
  else if (mat == 1) { W = Wk; bias = bk; }
  else               { W = Wv; bias = bv; }

  // A-layout fragments of x: lane holds row M=ln; elem e -> k = dc*32 + 16*hi + e
  const int row = tile * 16 + ln;
  const float* xr = x + (size_t)row * DIN + 16 * hi;
  v16h xa[8];
#pragma unroll
  for (int dc = 0; dc < 8; ++dc) {
#pragma unroll
    for (int e = 0; e < 16; ++e) {
      int k = dc * 32 + e;
      xa[dc][e] = (_Float16)xr[k];
    }
  }

  // log2(e)/sqrt(d_k): moves softmax into base-2 domain
  const float qscale = (mat == 0) ? (1.4426950408889634f * rsqrtf((float)DOUT)) : 1.0f;

#pragma unroll 1
  for (int nt = 0; nt < 8; ++nt) {
    const int col = nt * 16 + ln;
    v8f acc = {};
#pragma unroll
    for (int dc = 0; dc < 8; ++dc) {
      // B-layout of W: lane holds col n=col; elem e -> k = dc*32 + 16*hi + e
      const float* wp = W + (size_t)(dc * 32 + 16 * hi) * DOUT + col;
      v16h wb;
#pragma unroll
      for (int e = 0; e < 16; ++e) wb[e] = (_Float16)wp[(size_t)e * DOUT];
      acc = wmma_f16(xa[dc], wb, acc);
    }
    const float b = bias[col];
#pragma unroll
    for (int r = 0; r < 8; ++r) {        // C/D layout: row M = r + 8*hi, col N = ln
      float v = (acc[r] + b) * qscale;
      const int orow = tile * 16 + r + 8 * hi;
      if (mat == 2)      Vt[(size_t)col * N_ROWS + orow] = (_Float16)v;
      else if (mat == 1) Kh[(size_t)orow * DOUT + col]   = (_Float16)v;
      else               Qh[(size_t)orow * DOUT + col]   = (_Float16)v;
    }
  }
}

// ---------------------------------------------------------------------------
// Kernel 2: fused graph-masked attention + ELU, unnormalized-exp + split-K.
// Sᵀ trick: T = K_tile @ Qᵀ puts scores directly in the A-fragment layout
// needed for P@V — no LDS round trip, no shuffles, no rescale in the loop.
// Block = 256 threads = 8 waves = 2 query tiles x 4 key splits.
// Partial (O, l) merged with one-time ds_add_f32 reductions.
// ---------------------------------------------------------------------------
__global__ void __launch_bounds__(256) flash_gat_kernel(
    const _Float16* __restrict__ Qh, const _Float16* __restrict__ Kh,
    const _Float16* __restrict__ Vt, const int* __restrict__ adj,
    float* __restrict__ out)
{
  __shared__ float ored[2][16][DOUT];   // merged O partials per query tile (16 KB)
  __shared__ float lred[2][16];         // merged softmax denominators

  const int lane  = threadIdx.x & 31;
  const int wid   = threadIdx.x >> 5;
  const int hi    = lane >> 4;
  const int ln    = lane & 15;
  const int qt_l  = wid >> 2;                 // 0..1  local query tile
  const int split = wid & 3;                  // 0..3  key split
  const int qtile = blockIdx.x * 2 + qt_l;    // 0..511
  const int qbase = qtile * 16;
  const int kb0   = split * KSPAN;
  const int kb1   = kb0 + KSPAN;

  // zero the LDS reduction buffers
  for (int i = threadIdx.x; i < 2 * 16 * DOUT; i += 256)
    ((float*)ored)[i] = 0.0f;
  if (threadIdx.x < 32) ((float*)lred)[threadIdx.x] = 0.0f;
  __syncthreads();

  // Q B-fragments (Qᵀ for T = K @ Qᵀ): lane holds q col = ln; elem e -> d = dc*32 + 16*hi + e
  v16h qb[4];
  {
    const _Float16* qp = Qh + (size_t)(qbase + ln) * DOUT + 16 * hi;
#pragma unroll
    for (int dc = 0; dc < 4; ++dc) qb[dc] = *(const v16h*)(qp + dc * 32);
  }

  v8f o[8] = {};        // O partial: C-layout row q = r + 8*hi, col d = dt*16 + ln
  float lsum = 0.0f;    // per-lane partial denominator for q = ln

#pragma unroll 1
  for (int kb = kb0; kb < kb1; kb += 32) {
    if (kb + 32 < kb1)  // prefetch next adj block (global_prefetch_b8)
      __builtin_prefetch(adj + (size_t)(qbase + ln) * N_ROWS + kb + 32 + 8 * hi, 0, 1);

    // ---- Tᵀ scores: T_kt = K_tile(16x128) @ Qᵀ(128x16); lane = (key r+8*hi, q ln) ----
    v8f t0 = {}, t1 = {};
#pragma unroll
    for (int dc = 0; dc < 4; ++dc) {
      // A-layout of K: lane holds key row; elem e -> d = dc*32 + 16*hi + e
      const _Float16* kp0 = Kh + (size_t)(kb + ln) * DOUT + dc * 32 + 16 * hi;
      const _Float16* kp1 = Kh + (size_t)(kb + 16 + ln) * DOUT + dc * 32 + 16 * hi;
      v8h l0 = *(const v8h*)kp0, h0 = *(const v8h*)(kp0 + 8);
      v8h l1 = *(const v8h*)kp1, h1 = *(const v8h*)(kp1 + 8);
      v16h ka0, ka1;
#pragma unroll
      for (int e = 0; e < 8; ++e) {
        ka0[e] = l0[e]; ka0[e + 8] = h0[e];
        ka1[e] = l1[e]; ka1[e + 8] = h1[e];
      }
      t0 = wmma_f16(ka0, qb[dc], t0);
      t1 = wmma_f16(ka1, qb[dc], t1);
    }

    // ---- adjacency row for q = ln: 4 x b128 loads cover keys kb+8*hi+{0..7,16..23} ----
    const int* ap = adj + (size_t)(qbase + ln) * N_ROWS + kb + 8 * hi;
    v4i a00 = *(const v4i*)(ap);
    v4i a01 = *(const v4i*)(ap + 4);
    v4i a10 = *(const v4i*)(ap + 16);
    v4i a11 = *(const v4i*)(ap + 20);

    // ---- P = adj ? exp2(T) : 0 ; build P A-fragment in registers (Sᵀ trick) ----
    v16h pa;
#pragma unroll
    for (int r = 0; r < 8; ++r) {
      const int m0 = (r < 4) ? a00[r] : a01[r - 4];
      const int m1 = (r < 4) ? a10[r] : a11[r - 4];
      const float p0 = (m0 != 0) ? exp2f(t0[r]) : 0.0f;
      const float p1 = (m1 != 0) ? exp2f(t1[r]) : 0.0f;
      lsum += p0 + p1;
      pa[r]     = (_Float16)p0;   // A elem e<8  <-> key = kb + 8*hi + e      = t0[e]
      pa[r + 8] = (_Float16)p1;   // A elem e>=8 <-> key = kb + 16 + 8*hi + (e-8) = t1[e-8]
    }

    // ---- O += P(16x32) @ V(32x128): Vt transposed -> contiguous B-fragments ----
#pragma unroll
    for (int dt = 0; dt < 8; ++dt) {
      const _Float16* vp = Vt + (size_t)(dt * 16 + ln) * N_ROWS + kb + 8 * hi;
      v8h vl = *(const v8h*)vp, vh = *(const v8h*)(vp + 16);
      v16h vb = __builtin_shufflevector(vl, vh, 0, 1, 2, 3, 4, 5, 6, 7, 8, 9, 10, 11, 12, 13, 14, 15);
      o[dt] = wmma_f16(pa, vb, o[dt]);
    }
  }

  // ---- merge split-K partials: plain sums (no max-rescale needed) ----
#pragma unroll
  for (int r = 0; r < 8; ++r)
#pragma unroll
    for (int dt = 0; dt < 8; ++dt)
      atomicAdd(&ored[qt_l][r + 8 * hi][dt * 16 + ln], o[dt][r]);   // ds_add_f32
  atomicAdd(&lred[qt_l][ln], lsum);   // lanes ln and ln+16 both contribute
  __syncthreads();

  // ---- epilogue (one wave per query tile): normalize, ELU, store f32 ----
  if (split == 0) {
#pragma unroll
    for (int r = 0; r < 8; ++r) {
      const float l   = lred[qt_l][r + 8 * hi];
      const float inv = (l > 0.0f) ? (1.0f / l) : 0.0f;
      const int orow  = qbase + r + 8 * hi;
#pragma unroll
      for (int dt = 0; dt < 8; ++dt) {
        float v = ored[qt_l][r + 8 * hi][dt * 16 + ln] * inv;
        v = (v > 0.0f) ? v : (__expf(v) - 1.0f);
        out[(size_t)orow * DOUT + dt * 16 + ln] = v;
      }
    }
  }
}

// ---------------------------------------------------------------------------
extern "C" void kernel_launch(void* const* d_in, const int* in_sizes, int n_in,
                              void* d_out, int out_size, void* d_ws, size_t ws_size,
                              hipStream_t stream) {
  const float* x   = (const float*)d_in[0];
  const int*   adj = (const int*)d_in[1];
  const float* Wq  = (const float*)d_in[2];
  const float* bq  = (const float*)d_in[3];
  const float* Wk  = (const float*)d_in[4];
  const float* bk  = (const float*)d_in[5];
  const float* Wv  = (const float*)d_in[6];
  const float* bv  = (const float*)d_in[7];
  float* out = (float*)d_out;

  // workspace: Qh (2MB) | Kh (2MB) | Vt (2MB), all f16 (L2-resident)
  _Float16* Qh = (_Float16*)d_ws;
  _Float16* Kh = Qh + (size_t)N_ROWS * DOUT;
  _Float16* Vt = Kh + (size_t)N_ROWS * DOUT;

  // 1536 wave-jobs (512 row-tiles x 3 matrices), 8 waves/block
  qkv_proj_kernel<<<192, 256, 0, stream>>>(x, Wq, bq, Wk, bk, Wv, bv, Qh, Kh, Vt);
  // 512 query tiles x 4 key-splits = 2048 waves; 8 waves/block -> 256 blocks
  flash_gat_kernel<<<256, 256, 0, stream>>>(Qh, Kh, Vt, adj, out);
}
